// GatedFusionMambaBlock_76338748719293
// MI455X (gfx1250) — matrix operand order repaired and checked by reading, not compiled, not run on hardware
//
#include <hip/hip_runtime.h>
#include <cstdint>
#include <cstddef>

#define D_MODEL   512
#define FPQ       15
#define SEQ       2048
#define BSZ       2
#define D_STATE   16
#define D_CONV    4
#define HEADDIM   64
#define NHEADS    16
#define D_INNER   1024
#define CONV_DIM  (D_INNER + 2 * D_STATE)            // 1056
#define D_IN_PROJ (2 * D_INNER + 2 * D_STATE + NHEADS) // 2096
#define D_FF      2048
#define ROWS      (BSZ * SEQ)                        // 4096
#define EPS_GATE  0.01f

typedef __bf16 bf16_t;
typedef bf16_t v16bf __attribute__((ext_vector_type(16)));
typedef float  v8f   __attribute__((ext_vector_type(8)));

union FragBF { v16bf v; uint4 q[2]; };

__device__ __forceinline__ uint16_t f2bf(float f) {
  union { float f; uint32_t u; } v; v.f = f;
  uint32_t r = v.u + 0x7fffu + ((v.u >> 16) & 1u);
  return (uint16_t)(r >> 16);
}
__device__ __forceinline__ float siluf(float x) { return x / (1.0f + __expf(-x)); }

// CDNA5 async LDS-direct load (ASYNCcnt-tracked). LDS dest = low 32 bits of
// the generic shared-memory address (aperture bits live in [63:32]).
__device__ __forceinline__ void async_load_b128(uint32_t lds_off,
                                                const uint16_t* g) {
  asm volatile("global_load_async_to_lds_b128 %0, %1, off"
               :
               : "v"(lds_off), "v"(g)
               : "memory");
}
__device__ __forceinline__ void wait_async0() {
  asm volatile("s_wait_asynccnt 0" ::: "memory");
}

// ---------------------------------------------------------------------------
// Casts: plain (activations) and zero-padded (weights -> branch-free GEMM)
// ---------------------------------------------------------------------------
__global__ __launch_bounds__(256) void k_cast_bf16(const float* __restrict__ src,
                                                   uint16_t* __restrict__ dst, int n) {
  int i = blockIdx.x * 256 + threadIdx.x;
  if (i < n) dst[i] = f2bf(src[i]);
}

__global__ __launch_bounds__(256) void k_cast_pad(const float* __restrict__ src,
                                                  uint16_t* __restrict__ dst,
                                                  int nvalid, int ntotal) {
  int i = blockIdx.x * 256 + threadIdx.x;
  if (i < ntotal) dst[i] = (i < nvalid) ? f2bf(src[i]) : (uint16_t)0;
}

// ---------------------------------------------------------------------------
// RMSNorm over D (one block per row). outF / outB may be null.
// ---------------------------------------------------------------------------
__global__ __launch_bounds__(256) void k_rmsnorm(const float* __restrict__ x,
                                                 const float* __restrict__ w,
                                                 float* __restrict__ outF,
                                                 uint16_t* __restrict__ outB,
                                                 int D, float eps) {
  __shared__ float red[256];
  int row = blockIdx.x;
  const float* xr = x + (size_t)row * D;
  float s = 0.f;
  for (int d = threadIdx.x; d < D; d += 256) { float v = xr[d]; s += v * v; }
  red[threadIdx.x] = s; __syncthreads();
  for (int off = 128; off; off >>= 1) {
    if (threadIdx.x < off) red[threadIdx.x] += red[threadIdx.x + off];
    __syncthreads();
  }
  float scale = rsqrtf(red[0] / (float)D + eps);
  for (int d = threadIdx.x; d < D; d += 256) {
    float v = xr[d] * scale * w[d];
    if (outF) outF[(size_t)row * D + d] = v;
    if (outB) outB[(size_t)row * D + d] = f2bf(v);
  }
}

// ---------------------------------------------------------------------------
// Gated fusion: one block per chunk of 16 rows (15 frames + 1 query).
// ---------------------------------------------------------------------------
__global__ __launch_bounds__(256) void k_gated_fusion(float* __restrict__ h,
                                                      const float* __restrict__ wfc,
                                                      const float* __restrict__ wfcb,
                                                      const float* __restrict__ gfc,
                                                      const float* __restrict__ gfcb) {
  __shared__ float q[D_MODEL];
  __shared__ float red16[16];
  __shared__ float alpha[16];
  __shared__ float gateS;
  int chunk = blockIdx.x;
  int base = chunk * 16;
  int qrow = base + FPQ;
  const int tid = threadIdx.x;
  for (int d = tid; d < D_MODEL; d += 256) q[d] = h[(size_t)qrow * D_MODEL + d];
  __syncthreads();

  int f = tid >> 4, l16 = tid & 15;
  const float* wp = (f < FPQ) ? (wfc + (size_t)f * D_MODEL) : gfc;
  float p = 0.f;
  for (int d = l16; d < D_MODEL; d += 16) p += q[d] * wp[d];
  for (int off = 8; off; off >>= 1) p += __shfl_xor(p, off, 16);
  if (l16 == 0) red16[f] = p;
  __syncthreads();

  if (tid == 0) {
    float m = -1e30f;
    for (int i = 0; i < FPQ; ++i) m = fmaxf(m, red16[i] + wfcb[i]);
    float s = 0.f;
    for (int i = 0; i < FPQ; ++i) { float e = __expf(red16[i] + wfcb[i] - m); alpha[i] = e; s += e; }
    float inv = 1.0f / s;
    for (int i = 0; i < FPQ; ++i) alpha[i] *= inv;
    float g = 1.0f / (1.0f + __expf(-(red16[FPQ] + gfcb[0])));
    gateS = g * (1.0f - 2.0f * EPS_GATE) + EPS_GATE;
  }
  __syncthreads();

  float g = gateS;
  for (int d = tid; d < D_MODEL; d += 256) {
    float agg = 0.f;
#pragma unroll
    for (int i = 0; i < FPQ; ++i) agg += h[(size_t)(base + i) * D_MODEL + d] * alpha[i];
    h[(size_t)qrow * D_MODEL + d] = q[d] * (1.0f - g) + agg * g;
  }
}

// ---------------------------------------------------------------------------
// bf16 WMMA GEMM: C[M,N] = A[M,K] * B[N,K]^T (row-major, K contiguous).
// B rows are zero-padded to a multiple of 128 -> branch-free K-loop.
// Double-buffered LDS filled by async LDS-direct loads (ASYNCcnt).
// OP: 0 = store f32, 1 = accumulate into f32, 2 = bias + silu -> bf16
// ---------------------------------------------------------------------------
template<int OP>
__global__ __launch_bounds__(256) void k_gemm_bf16(const uint16_t* __restrict__ A,
                                                   const uint16_t* __restrict__ B,
                                                   float* __restrict__ C,
                                                   uint16_t* __restrict__ Cb,
                                                   const float* __restrict__ bias,
                                                   int M, int N, int K) {
  constexpr int LD = 40;   // padded row stride (bf16 units) -> 80B, bank stagger
  __shared__ __align__(16) uint16_t As[2][128 * LD];
  __shared__ __align__(16) uint16_t Bs[2][128 * LD];
  const int tid  = threadIdx.x;
  const int bm   = blockIdx.y * 128;
  const int bn   = blockIdx.x * 128;
  const int lane = tid & 31;
  const int wave = tid >> 5;
  const int wm   = (wave & 3) * 32;   // 4 waves along M
  const int wn   = (wave >> 2) * 64;  // 2 waves along N

  v8f acc[2][4];
  v8f zero8 = {};
#pragma unroll
  for (int i = 0; i < 2; ++i)
#pragma unroll
    for (int j = 0; j < 4; ++j) acc[i][j] = zero8;

  // Per-thread tile-fill assignment: 2 x 16B chunks each for A and B.
  const int row0 = tid >> 2;                 // 0..63
  const int row1 = row0 + 64;                // 64..127
  const int c8   = (tid & 3) * 8;            // bf16 col offset 0/8/16/24
  const uint16_t* gA0 = A + (size_t)(bm + row0) * K + c8;
  const uint16_t* gA1 = A + (size_t)(bm + row1) * K + c8;
  const uint16_t* gB0 = B + (size_t)(bn + row0) * K + c8;
  const uint16_t* gB1 = B + (size_t)(bn + row1) * K + c8;
  const uint32_t lA0[2] = {(uint32_t)(uintptr_t)(&As[0][row0 * LD + c8]),
                           (uint32_t)(uintptr_t)(&As[1][row0 * LD + c8])};
  const uint32_t lA1[2] = {(uint32_t)(uintptr_t)(&As[0][row1 * LD + c8]),
                           (uint32_t)(uintptr_t)(&As[1][row1 * LD + c8])};
  const uint32_t lB0[2] = {(uint32_t)(uintptr_t)(&Bs[0][row0 * LD + c8]),
                           (uint32_t)(uintptr_t)(&Bs[1][row0 * LD + c8])};
  const uint32_t lB1[2] = {(uint32_t)(uintptr_t)(&Bs[0][row1 * LD + c8]),
                           (uint32_t)(uintptr_t)(&Bs[1][row1 * LD + c8])};

  const int kbA  = (lane < 16) ? 0 : 8;   // A: K chunks {kbA..+8, kbA+16..+8}
  const int kcB  = (lane < 16) ? 0 : 8;   // B: K chunks {kcB..+8, kcB+16..+8}
  const int rsel = lane & 15;

  const int nk = K >> 5;
  // prologue: fill buffer 0 with tile 0
  async_load_b128(lA0[0], gA0);
  async_load_b128(lA1[0], gA1);
  async_load_b128(lB0[0], gB0);
  async_load_b128(lB1[0], gB1);

  for (int it = 0; it < nk; ++it) {
    const int cur = it & 1;
    wait_async0();        // own tile-it loads landed in LDS
    __syncthreads();      // all waves' loads landed; all done reading other buf
    if (it + 1 < nk) {
      const int nxt = cur ^ 1;
      const int koff = (it + 1) << 5;
      async_load_b128(lA0[nxt], gA0 + koff);
      async_load_b128(lA1[nxt], gA1 + koff);
      async_load_b128(lB0[nxt], gB0 + koff);
      async_load_b128(lB1[nxt], gB1 + koff);
    }

    const uint16_t* as = As[cur];
    const uint16_t* bs = Bs[cur];
    FragBF af[2], bfr[4];
#pragma unroll
    for (int mt = 0; mt < 2; ++mt) {
      int r = wm + mt * 16 + rsel;
      af[mt].q[0] = *(const uint4*)(as + r * LD + kbA);
      af[mt].q[1] = *(const uint4*)(as + r * LD + kbA + 16);
    }
#pragma unroll
    for (int nt = 0; nt < 4; ++nt) {
      int r = wn + nt * 16 + rsel;
      bfr[nt].q[0] = *(const uint4*)(bs + r * LD + kcB);
      bfr[nt].q[1] = *(const uint4*)(bs + r * LD + kcB + 16);
    }
#pragma unroll
    for (int mt = 0; mt < 2; ++mt)
#pragma unroll
      for (int nt = 0; nt < 4; ++nt)
        acc[mt][nt] = __builtin_amdgcn_wmma_f32_16x16x32_bf16(
            false, af[mt].v, false, bfr[nt].v, (short)0, acc[mt][nt], false, false);
  }

  const int rbase = (lane < 16) ? 0 : 8;
#pragma unroll
  for (int mt = 0; mt < 2; ++mt)
#pragma unroll
    for (int nt = 0; nt < 4; ++nt) {
      int col = bn + wn + nt * 16 + rsel;
      if (col >= N) continue;
#pragma unroll
      for (int r = 0; r < 8; ++r) {
        int row = bm + wm + mt * 16 + rbase + r;
        float v = acc[mt][nt][r];
        size_t o = (size_t)row * N + col;
        if (OP == 0) C[o] = v;
        else if (OP == 1) C[o] += v;
        else if (OP == 2) { v += bias[col]; Cb[o] = f2bf(siluf(v)); }
      }
    }
}

// ---------------------------------------------------------------------------
// Causal (dir=0) / anti-causal (dir=1) depthwise conv over xBC + SiLU.
// ---------------------------------------------------------------------------
__global__ __launch_bounds__(256) void k_conv_silu(const float* __restrict__ zx,
                                                   const float* __restrict__ convw,
                                                   const float* __restrict__ convb,
                                                   float* __restrict__ out, int dir) {
  int row = blockIdx.x;            // b*SEQ + t
  int b = row / SEQ, t = row % SEQ;
  for (int c = threadIdx.x; c < CONV_DIM; c += 256) {
    float acc = convb[c];
#pragma unroll
    for (int k = 0; k < D_CONV; ++k) {
      int tt = (dir == 0) ? (t + k - (D_CONV - 1)) : (t + (D_CONV - 1) - k);
      if (tt >= 0 && tt < SEQ)
        acc += zx[((size_t)(b * SEQ + tt)) * D_IN_PROJ + D_INNER + c] * convw[c * D_CONV + k];
    }
    out[(size_t)row * CONV_DIM + c] = siluf(acc);
  }
}

// ---------------------------------------------------------------------------
// dt = softplus(dt_raw + dtbias);  dA = exp(dt * (-exp(Alog)))
// ---------------------------------------------------------------------------
__global__ __launch_bounds__(256) void k_dt_dA(const float* __restrict__ zx,
                                               const float* __restrict__ dtbias,
                                               const float* __restrict__ Alog,
                                               float* __restrict__ dt,
                                               float* __restrict__ dA) {
  int i = blockIdx.x * 256 + threadIdx.x;        // ROWS*NHEADS
  if (i >= ROWS * NHEADS) return;
  int row = i / NHEADS, h = i % NHEADS;
  float x = zx[(size_t)row * D_IN_PROJ + 2 * D_INNER + 2 * D_STATE + h] + dtbias[h];
  float dtv = (x > 20.f) ? x : log1pf(__expf(x));
  dt[i] = dtv;
  dA[i] = __expf(dtv * (-__expf(Alog[h])));
}

// ---------------------------------------------------------------------------
// Dual-direction selective scan. 64 blocks: dir = blk>>5, (b,head) = blk&31.
// ---------------------------------------------------------------------------
__global__ __launch_bounds__(64) void k_scan(const float* __restrict__ xBC0,
                                             const float* __restrict__ xBC1,
                                             const float* __restrict__ dt0,
                                             const float* __restrict__ dA0,
                                             const float* __restrict__ dt1,
                                             const float* __restrict__ dA1,
                                             const float* __restrict__ fD,
                                             const float* __restrict__ bD,
                                             float* __restrict__ y0,
                                             float* __restrict__ y1) {
  int blk = blockIdx.x;
  int dir = blk >> 5;
  int bh  = blk & 31;
  int b = bh >> 4, head = bh & 15;
  int p = threadIdx.x;                       // 0..63
  const float* xbc = dir ? xBC1 : xBC0;
  const float* dtp = dir ? dt1 : dt0;
  const float* dap = dir ? dA1 : dA0;
  float* yout      = dir ? y1 : y0;
  float dcoef = (dir ? bD : fD)[head];

  float h[D_STATE];
#pragma unroll
  for (int n = 0; n < D_STATE; ++n) h[n] = 0.f;

  for (int step = 0; step < SEQ; ++step) {
    int t = dir ? (SEQ - 1 - step) : step;
    size_t row = (size_t)(b * SEQ + t);
    const float* xr = xbc + row * CONV_DIM;
    float x = xr[head * HEADDIM + p];
    const float4* B4 = (const float4*)(xr + D_INNER);
    const float4* C4 = (const float4*)(xr + D_INNER + D_STATE);
    float dtv = dtp[row * NHEADS + head];
    float dAv = dap[row * NHEADS + head];
    float u = dtv * x;
    float acc = 0.f;
#pragma unroll
    for (int q = 0; q < 4; ++q) {
      float4 Bv = B4[q], Cv = C4[q];
      h[q*4+0] = dAv * h[q*4+0] + u * Bv.x;  acc += h[q*4+0] * Cv.x;
      h[q*4+1] = dAv * h[q*4+1] + u * Bv.y;  acc += h[q*4+1] * Cv.y;
      h[q*4+2] = dAv * h[q*4+2] + u * Bv.z;  acc += h[q*4+2] * Cv.z;
      h[q*4+3] = dAv * h[q*4+3] + u * Bv.w;  acc += h[q*4+3] * Cv.w;
    }
    yout[row * D_INNER + head * HEADDIM + p] = acc + dcoef * x;
  }
}

// ---------------------------------------------------------------------------
// g = y * silu(z); rmsnorm over D_INNER; write bf16. One block per row.
// ---------------------------------------------------------------------------
__global__ __launch_bounds__(256) void k_gate_norm(const float* __restrict__ y,
                                                   const float* __restrict__ zx,
                                                   const float* __restrict__ normw,
                                                   uint16_t* __restrict__ out) {
  __shared__ float red[256];
  int row = blockIdx.x;
  const float* yr = y + (size_t)row * D_INNER;
  const float* zr = zx + (size_t)row * D_IN_PROJ;     // z region = first D_INNER
  float g[4]; float s = 0.f;
#pragma unroll
  for (int i = 0; i < 4; ++i) {
    int d = threadIdx.x + i * 256;
    float v = yr[d] * siluf(zr[d]);
    g[i] = v; s += v * v;
  }
  red[threadIdx.x] = s; __syncthreads();
  for (int off = 128; off; off >>= 1) {
    if (threadIdx.x < off) red[threadIdx.x] += red[threadIdx.x + off];
    __syncthreads();
  }
  float scale = rsqrtf(red[0] / (float)D_INNER + 1e-5f);
#pragma unroll
  for (int i = 0; i < 4; ++i) {
    int d = threadIdx.x + i * 256;
    out[(size_t)row * D_INNER + d] = f2bf(g[i] * scale * normw[d]);
  }
}

__global__ __launch_bounds__(256) void k_add_resid(float* __restrict__ a,
                                                   const float* __restrict__ r, int n) {
  int i = blockIdx.x * 256 + threadIdx.x;
  if (i < n) a[i] += r[i];
}

__global__ __launch_bounds__(256) void k_final(const float* __restrict__ fc2out,
                                               const float* __restrict__ fc2b,
                                               const float* __restrict__ rm,
                                               const float* __restrict__ resid,
                                               float* __restrict__ out, int n) {
  int i = blockIdx.x * 256 + threadIdx.x;
  if (i < n) out[i] = fc2out[i] + fc2b[i & (D_MODEL - 1)] + rm[i] + resid[i];
}

// ---------------------------------------------------------------------------
extern "C" void kernel_launch(void* const* d_in, const int* in_sizes, int n_in,
                              void* d_out, int out_size, void* d_ws, size_t ws_size,
                              hipStream_t stream) {
  (void)in_sizes; (void)n_in; (void)out_size; (void)ws_size;
  const float* hidden  = (const float*)d_in[0];
  const float* norm1_w = (const float*)d_in[1];
  const float* wfc_w   = (const float*)d_in[2];
  const float* wfc_b   = (const float*)d_in[3];
  const float* gfc_w   = (const float*)d_in[4];
  const float* gfc_b   = (const float*)d_in[5];
  const float* norm2_w = (const float*)d_in[6];
  const float* fc1_w   = (const float*)d_in[7];
  const float* fc1_b   = (const float*)d_in[8];
  const float* fc2_w   = (const float*)d_in[9];
  const float* fc2_b   = (const float*)d_in[10];
  const float* inproj[2]  = {(const float*)d_in[11], (const float*)d_in[19]};
  const float* convw[2]   = {(const float*)d_in[12], (const float*)d_in[20]};
  const float* convb[2]   = {(const float*)d_in[13], (const float*)d_in[21]};
  const float* dtbias[2]  = {(const float*)d_in[14], (const float*)d_in[22]};
  const float* Alog[2]    = {(const float*)d_in[15], (const float*)d_in[23]};
  const float* Dp[2]      = {(const float*)d_in[16], (const float*)d_in[24]};
  const float* normw[2]   = {(const float*)d_in[17], (const float*)d_in[25]};
  const float* outproj[2] = {(const float*)d_in[18], (const float*)d_in[26]};
  float* out = (float*)d_out;

  // ---- workspace carve ----
  size_t off = 0;
  auto alloc = [&](size_t bytes) -> void* {
    void* p = (char*)d_ws + off;
    off += (bytes + 255) & ~(size_t)255;
    return p;
  };
  const int NPAD_IN = ((D_IN_PROJ + 127) / 128) * 128;   // 2176 (zero-padded rows)
  float*    h_f32   = (float*)alloc((size_t)ROWS * D_MODEL * 4);
  uint16_t* h_bf16  = (uint16_t*)alloc((size_t)ROWS * D_MODEL * 2);
  uint16_t* inpW[2]  = {(uint16_t*)alloc((size_t)NPAD_IN * D_MODEL * 2),
                        (uint16_t*)alloc((size_t)NPAD_IN * D_MODEL * 2)};
  uint16_t* outW[2]  = {(uint16_t*)alloc((size_t)D_MODEL * D_INNER * 2),
                        (uint16_t*)alloc((size_t)D_MODEL * D_INNER * 2)};
  uint16_t* fc1W     = (uint16_t*)alloc((size_t)D_FF * D_MODEL * 2);
  uint16_t* fc2W     = (uint16_t*)alloc((size_t)D_MODEL * D_FF * 2);
  float* zx[2]  = {(float*)alloc((size_t)ROWS * D_IN_PROJ * 4),
                   (float*)alloc((size_t)ROWS * D_IN_PROJ * 4)};
  float* xbc[2] = {(float*)alloc((size_t)ROWS * CONV_DIM * 4),
                   (float*)alloc((size_t)ROWS * CONV_DIM * 4)};
  float* dtb[2] = {(float*)alloc((size_t)ROWS * NHEADS * 4),
                   (float*)alloc((size_t)ROWS * NHEADS * 4)};
  float* dab[2] = {(float*)alloc((size_t)ROWS * NHEADS * 4),
                   (float*)alloc((size_t)ROWS * NHEADS * 4)};
  float* ybuf[2] = {(float*)alloc((size_t)ROWS * D_INNER * 4),
                    (float*)alloc((size_t)ROWS * D_INNER * 4)};
  float* outsum = (float*)alloc((size_t)ROWS * D_MODEL * 4);
  // reuse: ygated bf16 in xbc buffers; h2norm bf16 in ybuf[0];
  //        ffn1 bf16 in zx[1]; fc2out f32 in zx[0].
  uint16_t* ygated[2] = {(uint16_t*)xbc[0], (uint16_t*)xbc[1]};
  uint16_t* h2norm = (uint16_t*)ybuf[0];
  uint16_t* ffn1   = (uint16_t*)zx[1];
  float*    fc2out = zx[0];

  const int NE = ROWS * D_MODEL;  // 2,097,152

  // 1) weights -> bf16 (zero-padded rows where N % 128 != 0)
  auto castp = [&](const float* s, uint16_t* d, int nvalid, int ntotal) {
    k_cast_pad<<<(ntotal + 255) / 256, 256, 0, stream>>>(s, d, nvalid, ntotal);
  };
  castp(inproj[0], inpW[0], D_IN_PROJ * D_MODEL, NPAD_IN * D_MODEL);
  castp(inproj[1], inpW[1], D_IN_PROJ * D_MODEL, NPAD_IN * D_MODEL);
  castp(outproj[0], outW[0], D_MODEL * D_INNER, D_MODEL * D_INNER);
  castp(outproj[1], outW[1], D_MODEL * D_INNER, D_MODEL * D_INNER);
  castp(fc1_w, fc1W, D_FF * D_MODEL, D_FF * D_MODEL);
  castp(fc2_w, fc2W, D_MODEL * D_FF, D_MODEL * D_FF);

  // 2) rmsnorm1 -> h_f32
  k_rmsnorm<<<ROWS, 256, 0, stream>>>(hidden, norm1_w, h_f32, nullptr, D_MODEL, 1e-6f);
  // 3) gated fusion (in place on h_f32)
  k_gated_fusion<<<BSZ * (SEQ / (FPQ + 1)), 256, 0, stream>>>(h_f32, wfc_w, wfc_b, gfc_w, gfc_b);
  // 4) h -> bf16
  k_cast_bf16<<<(NE + 255) / 256, 256, 0, stream>>>(h_f32, h_bf16, NE);

  // 5-6) in_proj GEMMs (per-token; sequence flip only matters for conv/scan)
  dim3 gIn((D_IN_PROJ + 127) / 128, ROWS / 128);
  k_gemm_bf16<0><<<gIn, 256, 0, stream>>>(h_bf16, inpW[0], zx[0], nullptr, nullptr,
                                          ROWS, D_IN_PROJ, D_MODEL);
  k_gemm_bf16<0><<<gIn, 256, 0, stream>>>(h_bf16, inpW[1], zx[1], nullptr, nullptr,
                                          ROWS, D_IN_PROJ, D_MODEL);

  // 7-10) conv+silu, dt/dA per direction
  for (int d = 0; d < 2; ++d) {
    k_conv_silu<<<ROWS, 256, 0, stream>>>(zx[d], convw[d], convb[d], xbc[d], d);
    k_dt_dA<<<(ROWS * NHEADS + 255) / 256, 256, 0, stream>>>(zx[d], dtbias[d], Alog[d],
                                                             dtb[d], dab[d]);
  }

  // 11) dual-direction scan
  k_scan<<<64, 64, 0, stream>>>(xbc[0], xbc[1], dtb[0], dab[0], dtb[1], dab[1],
                                Dp[0], Dp[1], ybuf[0], ybuf[1]);

  // 12-13) gate * silu(z) + rmsnorm -> bf16 (writes into xbc memory, reused)
  k_gate_norm<<<ROWS, 256, 0, stream>>>(ybuf[0], zx[0], normw[0], ygated[0]);
  k_gate_norm<<<ROWS, 256, 0, stream>>>(ybuf[1], zx[1], normw[1], ygated[1]);

  // 14-15) out_proj GEMMs (fwd stores, bwd accumulates)
  dim3 gOut((D_MODEL + 127) / 128, ROWS / 128);
  k_gemm_bf16<0><<<gOut, 256, 0, stream>>>(ygated[0], outW[0], outsum, nullptr, nullptr,
                                           ROWS, D_MODEL, D_INNER);
  k_gemm_bf16<1><<<gOut, 256, 0, stream>>>(ygated[1], outW[1], outsum, nullptr, nullptr,
                                           ROWS, D_MODEL, D_INNER);

  // 16) rm = out_fwd + out_bwd + residual (in place)
  k_add_resid<<<(NE + 255) / 256, 256, 0, stream>>>(outsum, hidden, NE);

  // 17) rmsnorm2 -> bf16 (into ybuf[0] memory, reused)
  k_rmsnorm<<<ROWS, 256, 0, stream>>>(outsum, norm2_w, nullptr, h2norm, D_MODEL, 1e-6f);

  // 18) fc1 with fused bias + silu -> bf16
  dim3 gF1((D_FF + 127) / 128, ROWS / 128);
  k_gemm_bf16<2><<<gF1, 256, 0, stream>>>(h2norm, fc1W, nullptr, ffn1, fc1_b,
                                          ROWS, D_FF, D_MODEL);
  // 19) fc2
  dim3 gF2((D_MODEL + 127) / 128, ROWS / 128);
  k_gemm_bf16<0><<<gF2, 256, 0, stream>>>(ffn1, fc2W, fc2out, nullptr, nullptr,
                                          ROWS, D_MODEL, D_FF);

  // 20) out = fc2out + fc2_b + rm + residual
  k_final<<<(NE + 255) / 256, 256, 0, stream>>>(fc2out, fc2_b, outsum, hidden, out, NE);
}
